// BaseSupertaggingModel_35759897706656
// MI455X (gfx1250) — matrix operand order repaired and checked by reading, not compiled, not run on hardware
//
#include <hip/hip_runtime.h>
#include <hip/hip_bf16.h>
#include <math.h>

// ---------------------------------------------------------------------------
// BERT-base forward (B=16,S=256,D=768,L=12,H=12,F=3072) + wordpiece mean +
// classifier MLP (768->1024->425).  Heavy GEMMs run on CDNA5 WMMA
// (v_wmma_f32_16x16x32_f16, wave32) with double-buffered
// global_load_async_to_lds_b128 staging (ASYNCcnt-pipelined).
// ---------------------------------------------------------------------------

typedef __attribute__((ext_vector_type(16))) _Float16 v16h;
typedef __attribute__((ext_vector_type(8)))  _Float16 v8h;
typedef __attribute__((ext_vector_type(8)))  float    v8f;

#define BB   16
#define SS   256
#define DD   768
#define HH   12
#define DHH  64
#define FF   3072
#define LL   12
#define WW   128
#define NCC  425
#define MM   (BB*SS)          // 4096 tokens
#define LN_EPS 1e-12f

// ------------------------------ block reduce -------------------------------
__device__ __forceinline__ float blk_sum(float v, float* red) {
    int t = threadIdx.x;
    red[t] = v; __syncthreads();
    #pragma unroll
    for (int s = 128; s > 0; s >>= 1) { if (t < s) red[t] += red[t + s]; __syncthreads(); }
    float r = red[0]; __syncthreads();
    return r;
}
__device__ __forceinline__ float blk_max(float v, float* red) {
    int t = threadIdx.x;
    red[t] = v; __syncthreads();
    #pragma unroll
    for (int s = 128; s > 0; s >>= 1) { if (t < s) red[t] = fmaxf(red[t], red[t + s]); __syncthreads(); }
    float r = red[0]; __syncthreads();
    return r;
}

// --------------------- async global -> LDS helpers -------------------------
// Copies 32 bytes (16 halves) per lane: two b128 async transfers.
// lds_byte: per-lane LDS byte address (low 32 bits of flat shared address).
__device__ __forceinline__ void async_cp32(unsigned int lds_byte, const _Float16* g) {
    asm volatile("global_load_async_to_lds_b128 %0, %1, off"
                 :: "v"(lds_byte), "v"(g) : "memory");
    asm volatile("global_load_async_to_lds_b128 %0, %1, off"
                 :: "v"(lds_byte + 16u), "v"(g + 8) : "memory");
}
__device__ __forceinline__ unsigned int lds_addr32(const void* p) {
    return (unsigned int)(unsigned long long)p;   // LDS aperture: low 32 = offset
}

// --------------------------- WMMA tiled GEMM -------------------------------
// C[z][m][n] = alpha * sum_k A[z][m][k] * Bt[z][n][k]  (+ bias[n])  + epilogue
// A: [M,K] f16 row-major, Bt: [N,K] f16 row-major (i.e. B transposed).
// Tile 128x128, K-step 32, 8 waves/block, wave owns 32x64 (2x4 wmma tiles).
// K must be a multiple of 32 (true for all uses: 64,256,768,1024,3072).

enum { EPI_F32 = 0, EPI_GELU16, EPI_RELU16, EPI_QK, EPI_VT, EPI_CTX };

#define TM   128
#define TN   128
#define LDT  40      // LDS row stride in halves (80B, 16B-aligned, conflict pad)

template <int EPI>
__global__ __launch_bounds__(256)
void gemm_wmma(const _Float16* __restrict__ A, const _Float16* __restrict__ Bt,
               const float* __restrict__ bias, void* __restrict__ C,
               int M, int N, int K,
               long sA, long sB, long sC, float alpha)
{
    __shared__ _Float16 lA[2][TM * LDT];
    __shared__ _Float16 lB[2][TN * LDT];

    const int z = blockIdx.z;
    A  += (long)z * sA;
    Bt += (long)z * sB;

    const int m0 = blockIdx.y * TM;
    const int n0 = blockIdx.x * TN;
    const int t    = threadIdx.x;
    const int lane = t & 31;
    const int wave = t >> 5;          // 0..7
    const int wm   = (wave & 3) * 32; // wave row offset in tile
    const int wn   = (wave >> 2) * 64;

    const v8f vzero = {0.f,0.f,0.f,0.f,0.f,0.f,0.f,0.f};
    v8f acc[2][4];
    #pragma unroll
    for (int i = 0; i < 2; ++i)
        #pragma unroll
        for (int j = 0; j < 4; ++j) acc[i][j] = vzero;

    // staging assignment: thread t owns 16 halves of one row
    const int lrow = t >> 1;          // 0..127
    const int lcol = (t & 1) * 16;    // 0 or 16 halves
    int gm = m0 + lrow; if (gm >= M) gm = M - 1;   // clamp: OOB rows feed only
    int gn = n0 + lrow; if (gn >= N) gn = N - 1;   // guarded-out C cells
    const _Float16* gA = A  + (long)gm * K + lcol;
    const _Float16* gB = Bt + (long)gn * K + lcol;
    const unsigned int la0 = lds_addr32(&lA[0][lrow * LDT + lcol]);
    const unsigned int la1 = lds_addr32(&lA[1][lrow * LDT + lcol]);
    const unsigned int lb0 = lds_addr32(&lB[0][lrow * LDT + lcol]);
    const unsigned int lb1 = lds_addr32(&lB[1][lrow * LDT + lcol]);

    const int nk = K >> 5;

    // prologue: tile 0 -> buffer 0
    async_cp32(la0, gA);
    async_cp32(lb0, gB);

    for (int i = 0; i < nk; ++i) {
        const int cur = i & 1;
        if (i + 1 < nk) {
            // issue tile i+1 into the other buffer, then wait for tile i
            async_cp32(cur ? la0 : la1, gA + (i + 1) * 32);
            async_cp32(cur ? lb0 : lb1, gB + (i + 1) * 32);
            asm volatile("s_wait_asynccnt 0x4" ::: "memory");
        } else {
            asm volatile("s_wait_asynccnt 0x0" ::: "memory");
        }
        __syncthreads();

        // ---- build fragments (ISA VGPR layouts) ----
        const int r   = lane & 15;
        const int kbA = (lane >> 4) * 8;    // A: interleaved K halves per lane group
        const int kbB = (lane >> 4) * 8;    // B: interleaved K halves per lane group

        v16h af[2], bf[4];
        #pragma unroll
        for (int ii = 0; ii < 2; ++ii) {
            const _Float16* p = &lA[cur][(wm + ii * 16 + r) * LDT + kbA];
            v8h lo = *(const v8h*)(p);
            v8h hi = *(const v8h*)(p + 16);
            #pragma unroll
            for (int e = 0; e < 8; ++e) { af[ii][e] = lo[e]; af[ii][8 + e] = hi[e]; }
        }
        #pragma unroll
        for (int j = 0; j < 4; ++j) {
            const _Float16* p = &lB[cur][(wn + j * 16 + r) * LDT + kbB];
            v8h lo = *(const v8h*)(p);
            v8h hi = *(const v8h*)(p + 16);
            #pragma unroll
            for (int e = 0; e < 8; ++e) { bf[j][e] = lo[e]; bf[j][8 + e] = hi[e]; }
        }

        #pragma unroll
        for (int ii = 0; ii < 2; ++ii)
            #pragma unroll
            for (int j = 0; j < 4; ++j)
                acc[ii][j] = __builtin_amdgcn_wmma_f32_16x16x32_f16(
                    false, af[ii], false, bf[j], (short)0, acc[ii][j], false, false);

        __syncthreads();   // buffer `cur` is re-filled next iteration
    }

    // ---- epilogue: C layout lanes 0-15 -> M=r, lanes 16-31 -> M=r+8 ----
    const int cn = lane & 15;
    const int rm = (lane >> 4) * 8;
    #pragma unroll
    for (int i = 0; i < 2; ++i) {
        #pragma unroll
        for (int j = 0; j < 4; ++j) {
            #pragma unroll
            for (int e = 0; e < 8; ++e) {
                int m = m0 + wm + i * 16 + rm + e;
                int n = n0 + wn + j * 16 + cn;
                if (m >= M || n >= N) continue;
                float v = acc[i][j][e] * alpha;
                if (bias) v += bias[n];
                if (EPI == EPI_F32) {
                    ((float*)C)[(long)z * sC + (long)m * N + n] = v;
                } else if (EPI == EPI_GELU16) {
                    v = 0.5f * v * (1.0f + erff(v * 0.70710678118654752f));
                    ((_Float16*)C)[(long)z * sC + (long)m * N + n] = (_Float16)v;
                } else if (EPI == EPI_RELU16) {
                    v = fmaxf(v, 0.0f);
                    ((_Float16*)C)[(long)z * sC + (long)m * N + n] = (_Float16)v;
                } else if (EPI == EPI_QK) {
                    int b = m >> 8, s = m & 255, h = n >> 6, dh = n & 63;
                    ((_Float16*)C)[(((long)(b * HH + h)) * SS + s) * DHH + dh] = (_Float16)v;
                } else if (EPI == EPI_VT) {
                    int b = m >> 8, s = m & 255, h = n >> 6, dh = n & 63;
                    ((_Float16*)C)[(((long)(b * HH + h)) * DHH + dh) * SS + s] = (_Float16)v;
                } else if (EPI == EPI_CTX) {
                    int b = z / HH, h = z % HH;      // m = s, n = dh
                    ((_Float16*)C)[((long)(b * SS + m)) * DD + h * DHH + n] = (_Float16)v;
                }
            }
        }
    }
}

// ----------------------- f32 [K,N] -> f16 [N,K] transpose ------------------
__global__ __launch_bounds__(256)
void transpose_to_f16(const float* __restrict__ in, _Float16* __restrict__ out,
                      int K, int N, long strideIn, long strideOut)
{
    __shared__ float tile[32][33];
    in  += (long)blockIdx.z * strideIn;
    out += (long)blockIdx.z * strideOut;
    int k0 = blockIdx.x * 32, n0 = blockIdx.y * 32;
    int tx = threadIdx.x & 31, ty = threadIdx.x >> 5;   // 32x8
    for (int i = ty; i < 32; i += 8) {
        int k = k0 + i, n = n0 + tx;
        tile[i][tx] = (k < K && n < N) ? in[(long)k * N + n] : 0.0f;
    }
    __syncthreads();
    for (int i = ty; i < 32; i += 8) {
        int n = n0 + i, k = k0 + tx;
        if (n < N && k < K) out[(long)n * K + k] = (_Float16)tile[tx][i];
    }
}

// ----------------------- embeddings + LayerNorm ----------------------------
__global__ __launch_bounds__(256)
void embed_ln_kernel(const int* __restrict__ ids,
                     const float* __restrict__ we, const float* __restrict__ pe,
                     const float* __restrict__ te,
                     const float* __restrict__ g,  const float* __restrict__ b,
                     float* __restrict__ h32, _Float16* __restrict__ h16)
{
    __shared__ float red[256];
    int row = blockIdx.x;            // token index 0..4095
    int s   = row & 255;
    int id  = ids[row];
    int t   = threadIdx.x;
    float v[3]; float sum = 0.f;
    #pragma unroll
    for (int i = 0; i < 3; ++i) {
        int d = t + i * 256;
        float x = we[(long)id * DD + d] + pe[(long)s * DD + d] + te[d];
        v[i] = x; sum += x;
    }
    float mean = blk_sum(sum, red) * (1.0f / DD);
    float var = 0.f;
    #pragma unroll
    for (int i = 0; i < 3; ++i) { float d = v[i] - mean; var += d * d; }
    var = blk_sum(var, red) * (1.0f / DD);
    float rstd = rsqrtf(var + LN_EPS);
    #pragma unroll
    for (int i = 0; i < 3; ++i) {
        int d = t + i * 256;
        float o = (v[i] - mean) * rstd * g[d] + b[d];
        h32[(long)row * DD + d] = o;
        h16[(long)row * DD + d] = (_Float16)o;
    }
}

// ----------------------- residual + LayerNorm ------------------------------
__global__ __launch_bounds__(256)
void resln_kernel(float* __restrict__ h32, const float* __restrict__ y32,
                  const float* __restrict__ g, const float* __restrict__ b,
                  _Float16* __restrict__ h16)
{
    __shared__ float red[256];
    int row = blockIdx.x;
    int t = threadIdx.x;
    float v[3]; float sum = 0.f;
    #pragma unroll
    for (int i = 0; i < 3; ++i) {
        int d = t + i * 256;
        float x = h32[(long)row * DD + d] + y32[(long)row * DD + d];
        v[i] = x; sum += x;
    }
    float mean = blk_sum(sum, red) * (1.0f / DD);
    float var = 0.f;
    #pragma unroll
    for (int i = 0; i < 3; ++i) { float d = v[i] - mean; var += d * d; }
    var = blk_sum(var, red) * (1.0f / DD);
    float rstd = rsqrtf(var + LN_EPS);
    #pragma unroll
    for (int i = 0; i < 3; ++i) {
        int d = t + i * 256;
        float o = (v[i] - mean) * rstd * g[d] + b[d];
        h32[(long)row * DD + d] = o;
        h16[(long)row * DD + d] = (_Float16)o;
    }
}

// ----------------------- masked softmax -> f16 probs -----------------------
__global__ __launch_bounds__(256)
void softmax_kernel(const float* __restrict__ sc, const int* __restrict__ mask,
                    _Float16* __restrict__ pr)
{
    __shared__ float red[256];
    long row = blockIdx.x;                  // b*H*S + h*S + q
    int  b   = (int)(row / (HH * SS));
    int  t   = threadIdx.x;
    float mb = (1.0f - (float)mask[b * SS + t]) * -1e9f;
    float x  = sc[row * SS + t] + mb;
    float mx = blk_max(x, red);
    float e  = __expf(x - mx);
    float sm = blk_sum(e, red);
    pr[row * SS + t] = (_Float16)(e / sm);
}

// ----------------------- wordpiece segment mean ----------------------------
__global__ __launch_bounds__(256)
void wordpiece_kernel(const float* __restrict__ h32, const int* __restrict__ wp,
                      _Float16* __restrict__ f0)
{
    __shared__ int sh[2];
    int row = blockIdx.x;
    int b = row >> 8, s = row & 255;
    if (threadIdx.x == 0) {
        int start = 0, cnt = 1;
        if (s < WW) {
            for (int j = 0; j < s; ++j) start += wp[b * WW + j];
            cnt = wp[b * WW + s];
        } else {
            start = s; cnt = 1;          // rows >= W keep original values
        }
        sh[0] = start; sh[1] = cnt;
    }
    __syncthreads();
    int start = sh[0], cnt = sh[1];
    float inv = 1.0f / (float)cnt;
    int t = threadIdx.x;
    #pragma unroll
    for (int i = 0; i < 3; ++i) {
        int d = t + i * 256;
        float acc = 0.f;
        for (int q = 0; q < cnt; ++q)
            acc += h32[((long)b * SS + start + q) * DD + d];
        f0[((long)b * SS + s) * DD + d] = (_Float16)(acc * inv);
    }
}

// ---------------------------------------------------------------------------
static inline dim3 gemm_grid(int M, int N, int batch) {
    return dim3((N + TN - 1) / TN, (M + TM - 1) / TM, batch);
}

extern "C" void kernel_launch(void* const* d_in, const int* in_sizes, int n_in,
                              void* d_out, int out_size, void* d_ws, size_t ws_size,
                              hipStream_t stream)
{
    (void)in_sizes; (void)n_in; (void)out_size; (void)ws_size;

    // ---- inputs (setup_inputs() dict insertion order, nested dicts in order)
    const int*   ids   = (const int*)  d_in[0];
    const int*   maskp = (const int*)  d_in[1];
    const int*   wp    = (const int*)  d_in[2];
    const float* wemb  = (const float*)d_in[3];
    const float* pemb  = (const float*)d_in[4];
    const float* temb  = (const float*)d_in[5];
    const float* emb_g = (const float*)d_in[6];
    const float* emb_b = (const float*)d_in[7];
    const float* wq    = (const float*)d_in[8];
    const float* bq    = (const float*)d_in[9];
    const float* wk    = (const float*)d_in[10];
    const float* bk    = (const float*)d_in[11];
    const float* wv    = (const float*)d_in[12];
    const float* bv    = (const float*)d_in[13];
    const float* wo    = (const float*)d_in[14];
    const float* bo    = (const float*)d_in[15];
    const float* ln1g  = (const float*)d_in[16];
    const float* ln1b  = (const float*)d_in[17];
    const float* w1l   = (const float*)d_in[18];
    const float* b1l   = (const float*)d_in[19];
    const float* w2l   = (const float*)d_in[20];
    const float* b2l   = (const float*)d_in[21];
    const float* ln2g  = (const float*)d_in[22];
    const float* ln2b  = (const float*)d_in[23];
    const float* w1c   = (const float*)d_in[24];
    const float* b1c   = (const float*)d_in[25];
    const float* w2c   = (const float*)d_in[26];
    const float* b2c   = (const float*)d_in[27];
    float* out = (float*)d_out;

    // ---- workspace carve ----
    char* ws = (char*)d_ws;
    size_t off = 0;
    auto carve = [&](size_t bytes) -> void* {
        void* p = ws + off;
        off += (bytes + 255) & ~(size_t)255;
        return p;
    };
    float*     h32    = (float*)    carve((size_t)MM * DD * 4);
    _Float16*  h16    = (_Float16*) carve((size_t)MM * DD * 2);
    _Float16*  q16    = (_Float16*) carve((size_t)MM * DD * 2);   // [B,H,S,DH]
    _Float16*  k16    = (_Float16*) carve((size_t)MM * DD * 2);   // [B,H,S,DH]
    _Float16*  vT16   = (_Float16*) carve((size_t)MM * DD * 2);   // [B,H,DH,S]
    float*     sc32   = (float*)    carve((size_t)BB * HH * SS * SS * 4);
    _Float16*  pr16   = (_Float16*) carve((size_t)BB * HH * SS * SS * 2);
    _Float16*  ctx16  = (_Float16*) carve((size_t)MM * DD * 2);
    float*     y32    = (float*)    carve((size_t)MM * DD * 4);
    _Float16*  ffn16  = (_Float16*) carve((size_t)MM * FF * 2);
    _Float16*  f016   = (_Float16*) carve((size_t)MM * DD * 2);
    _Float16*  r16    = (_Float16*) carve((size_t)MM * 1024 * 2);
    // transposed f16 weight staging (reused every layer)
    _Float16*  wqT  = (_Float16*) carve((size_t)DD * DD * 2);
    _Float16*  wkT  = (_Float16*) carve((size_t)DD * DD * 2);
    _Float16*  wvT  = (_Float16*) carve((size_t)DD * DD * 2);
    _Float16*  woT  = (_Float16*) carve((size_t)DD * DD * 2);
    _Float16*  w1T  = (_Float16*) carve((size_t)FF * DD * 2);     // [3072,768]
    _Float16*  w2T  = (_Float16*) carve((size_t)DD * FF * 2);     // [768,3072]
    _Float16*  w1cT = (_Float16*) carve((size_t)1024 * DD * 2);   // [1024,768]
    _Float16*  w2cT = (_Float16*) carve((size_t)NCC * 1024 * 2);  // [425,1024]

    // ---- embeddings + LN ----
    embed_ln_kernel<<<MM, 256, 0, stream>>>(ids, wemb, pemb, temb, emb_g, emb_b, h32, h16);

    const dim3 tb(256);
    for (int l = 0; l < LL; ++l) {
        const float* wq_l = wq + (size_t)l * DD * DD;
        const float* wk_l = wk + (size_t)l * DD * DD;
        const float* wv_l = wv + (size_t)l * DD * DD;
        const float* wo_l = wo + (size_t)l * DD * DD;
        const float* w1_l = w1l + (size_t)l * DD * FF;
        const float* w2_l = w2l + (size_t)l * FF * DD;

        dim3 tg_dd(24, 24, 1);
        transpose_to_f16<<<tg_dd, tb, 0, stream>>>(wq_l, wqT, DD, DD, 0, 0);
        transpose_to_f16<<<tg_dd, tb, 0, stream>>>(wk_l, wkT, DD, DD, 0, 0);
        transpose_to_f16<<<tg_dd, tb, 0, stream>>>(wv_l, wvT, DD, DD, 0, 0);
        transpose_to_f16<<<tg_dd, tb, 0, stream>>>(wo_l, woT, DD, DD, 0, 0);
        transpose_to_f16<<<dim3(24, 96, 1), tb, 0, stream>>>(w1_l, w1T, DD, FF, 0, 0);
        transpose_to_f16<<<dim3(96, 24, 1), tb, 0, stream>>>(w2_l, w2T, FF, DD, 0, 0);

        // Q/K/V projections (scatter to attention layouts)
        gemm_wmma<EPI_QK><<<gemm_grid(MM, DD, 1), tb, 0, stream>>>(
            h16, wqT, bq + (size_t)l * DD, q16, MM, DD, DD, 0, 0, 0, 1.0f);
        gemm_wmma<EPI_QK><<<gemm_grid(MM, DD, 1), tb, 0, stream>>>(
            h16, wkT, bk + (size_t)l * DD, k16, MM, DD, DD, 0, 0, 0, 1.0f);
        gemm_wmma<EPI_VT><<<gemm_grid(MM, DD, 1), tb, 0, stream>>>(
            h16, wvT, bv + (size_t)l * DD, vT16, MM, DD, DD, 0, 0, 0, 1.0f);

        // scores = Q @ K^T * 1/sqrt(DH)    (batched over B*H)
        gemm_wmma<EPI_F32><<<gemm_grid(SS, SS, BB * HH), tb, 0, stream>>>(
            q16, k16, (const float*)nullptr, sc32, SS, SS, DHH,
            (long)SS * DHH, (long)SS * DHH, (long)SS * SS, 0.125f);

        softmax_kernel<<<BB * HH * SS, 256, 0, stream>>>(sc32, maskp, pr16);

        // ctx = P @ V   (Bt = V^T already in [DH,S])
        gemm_wmma<EPI_CTX><<<gemm_grid(SS, DHH, BB * HH), tb, 0, stream>>>(
            pr16, vT16, (const float*)nullptr, ctx16, SS, DHH, SS,
            (long)SS * SS, (long)SS * DHH, 0, 1.0f);

        // output projection + residual LN1
        gemm_wmma<EPI_F32><<<gemm_grid(MM, DD, 1), tb, 0, stream>>>(
            ctx16, woT, bo + (size_t)l * DD, y32, MM, DD, DD, 0, 0, 0, 1.0f);
        resln_kernel<<<MM, 256, 0, stream>>>(h32, y32,
            ln1g + (size_t)l * DD, ln1b + (size_t)l * DD, h16);

        // FFN
        gemm_wmma<EPI_GELU16><<<gemm_grid(MM, FF, 1), tb, 0, stream>>>(
            h16, w1T, b1l + (size_t)l * FF, ffn16, MM, FF, DD, 0, 0, 0, 1.0f);
        gemm_wmma<EPI_F32><<<gemm_grid(MM, DD, 1), tb, 0, stream>>>(
            ffn16, w2T, b2l + (size_t)l * DD, y32, MM, DD, FF, 0, 0, 0, 1.0f);
        resln_kernel<<<MM, 256, 0, stream>>>(h32, y32,
            ln2g + (size_t)l * DD, ln2b + (size_t)l * DD, h16);
    }

    // ---- wordpiece segment mean ----
    wordpiece_kernel<<<MM, 256, 0, stream>>>(h32, wp, f016);

    // ---- classifier MLP ----
    transpose_to_f16<<<dim3(24, 32, 1), tb, 0, stream>>>(w1c, w1cT, DD, 1024, 0, 0);
    gemm_wmma<EPI_RELU16><<<gemm_grid(MM, 1024, 1), tb, 0, stream>>>(
        f016, w1cT, b1c, r16, MM, 1024, DD, 0, 0, 0, 1.0f);

    transpose_to_f16<<<dim3(32, 14, 1), tb, 0, stream>>>(w2c, w2cT, 1024, NCC, 0, 0);
    gemm_wmma<EPI_F32><<<gemm_grid(MM, NCC, 1), tb, 0, stream>>>(
        r16, w2cT, b2c, out, MM, NCC, 1024, 0, 0, 0, 1.0f);
}
